// Node_44968307589673
// MI455X (gfx1250) — matrix operand order repaired and checked by reading, not compiled, not run on hardware
//
#include <hip/hip_runtime.h>

// ---------------------------------------------------------------------------
// MI455X (gfx1250) fully fused SOM-encoder:
//   z = x W^T + b ; w = softmax(-cdist(z,P)/T) ; blended = w @ P
// All three GEMMs via v_wmma_f32_16x16x32_bf16 (wave32), f32 accum.
// One mega-kernel per 32-row block; x-tile and z-tile share one LDS buffer.
// ---------------------------------------------------------------------------

typedef __attribute__((ext_vector_type(8)))  __bf16 v8bf;
typedef __attribute__((ext_vector_type(16))) __bf16 v16bf;
typedef __attribute__((ext_vector_type(8)))  float  v8f;

#define BROWS 65536
#define DIM   512
#define NPROT 1024
#define ROWS  32
#define LSTR  (DIM + 8)   // LDS row stride (bank-conflict-free frag reads)

// ---- WMMA fragment loaders (ISA 7.12.2 layouts) ---------------------------
// A (16xK): lane l%16 = row; element i -> K = k0 + 8*half + (i<8 ? i : i+8)
static __device__ __forceinline__ v16bf frag_a(const __bf16* rowp, int k0, int half) {
  const __bf16* p = rowp + k0 + (half << 3);
  v8bf lo = *(const v8bf*)(p);
  v8bf hi = *(const v8bf*)(p + 16);
  v16bf f;
#pragma unroll
  for (int i = 0; i < 8; ++i) { f[i] = lo[i]; f[i + 8] = hi[i]; }
  return f;
}

// B (Kx16 stored as B^T row-major [n][k]): lane l%16 = col n;
// element i -> K = k0 + 8*half + (i<8 ? i : i+8)
static __device__ __forceinline__ v16bf frag_b(const __bf16* rowp, int k0, int half) {
  const __bf16* p = rowp + k0 + (half << 3);
  v8bf lo = *(const v8bf*)(p);
  v8bf hi = *(const v8bf*)(p + 16);
  v16bf f;
#pragma unroll
  for (int i = 0; i < 8; ++i) { f[i] = lo[i]; f[i + 8] = hi[i]; }
  return f;
}

// A fragment from f32 global memory (weights readback), converted to bf16.
static __device__ __forceinline__ v16bf frag_a_f32(const float* rowp, int k0, int half) {
  const float* p = rowp + k0 + (half << 3);
  float4 a = *(const float4*)(p);
  float4 b = *(const float4*)(p + 4);
  float4 c = *(const float4*)(p + 16);
  float4 d = *(const float4*)(p + 20);
  v16bf f;
  f[0]=(__bf16)a.x;  f[1]=(__bf16)a.y;  f[2]=(__bf16)a.z;  f[3]=(__bf16)a.w;
  f[4]=(__bf16)b.x;  f[5]=(__bf16)b.y;  f[6]=(__bf16)b.z;  f[7]=(__bf16)b.w;
  f[8]=(__bf16)c.x;  f[9]=(__bf16)c.y;  f[10]=(__bf16)c.z; f[11]=(__bf16)c.w;
  f[12]=(__bf16)d.x; f[13]=(__bf16)d.y; f[14]=(__bf16)d.z; f[15]=(__bf16)d.w;
  return f;
}

static __device__ __forceinline__ v8f wmma_bf16(v16bf a, v16bf b, v8f c) {
  return __builtin_amdgcn_wmma_f32_16x16x32_bf16(false, a, false, b, (short)0, c,
                                                 false, false);
}

static __device__ __forceinline__ v8f zero8() {
  v8f z;
#pragma unroll
  for (int i = 0; i < 8; ++i) z[i] = 0.f;
  return z;
}

// ---------------------------------------------------------------------------
// Kernel 1: prep — bf16 copies of W, P, P^T; p_sq[n] = ||P[n]||^2
// ---------------------------------------------------------------------------
__global__ __launch_bounds__(128) void prep_kernel(
    const float* __restrict__ W, const float* __restrict__ protos,
    __bf16* __restrict__ Wb, __bf16* __restrict__ Pb,
    __bf16* __restrict__ PTb, float* __restrict__ psq) {
  __shared__ float red[128];
  const int bid = blockIdx.x;
  const int tid = threadIdx.x;
  if (bid < NPROT) {
    const float* src = protos + (size_t)bid * DIM;
    float s = 0.f;
    for (int i = tid; i < DIM; i += 128) {
      float v = src[i];
      __bf16 bv = (__bf16)v;
      Pb[(size_t)bid * DIM + i]    = bv;   // [N][D] (GEMM2 B^T)
      PTb[(size_t)i * NPROT + bid] = bv;   // [D][N] (GEMM3 B^T)
      s += v * v;
    }
    red[tid] = s;
    __syncthreads();
#pragma unroll
    for (int off = 64; off > 0; off >>= 1) {
      if (tid < off) red[tid] += red[tid + off];
      __syncthreads();
    }
    if (tid == 0) psq[bid] = red[0];
  } else {
    const int r = bid - NPROT;
    const float* src = W + (size_t)r * DIM;
    for (int i = tid; i < DIM; i += 128)
      Wb[(size_t)r * DIM + i] = (__bf16)src[i];  // [D][D] (GEMM1 B^T = W)
  }
}

// ---------------------------------------------------------------------------
// Kernel 2: mega-fused. 32 rows per WG (grid 2048), 256 threads = 8 waves.
// Each wave handles ALL 32 rows (2 M-tiles) x 1/8 of columns -> B-fragment
// reuse across M, per-WG B traffic = W(0.5MB)+P(1MB)+P^T(1MB) from L2.
// ---------------------------------------------------------------------------
__global__ __launch_bounds__(256) void som_fused_kernel(
    const float* __restrict__ x, const __bf16* __restrict__ Wb,
    const float* __restrict__ bias,
    const __bf16* __restrict__ Pb, const __bf16* __restrict__ PTb,
    const float* __restrict__ psq, const float* __restrict__ temp_raw,
    float* __restrict__ out_blend, float* __restrict__ out_w) {

  __shared__ __bf16 buf[ROWS * LSTR];   // phase 0-1: x tile ; phase 2+: z tile
  __shared__ float  zsql[ROWS];
  __shared__ float  rowred[ROWS][8];
  __shared__ float  rstat[ROWS];        // row max, then 1/rowsum

  const int tid  = threadIdx.x;
  const int lane = tid & 31;
  const int wv   = tid >> 5;
  const int half = lane >> 4;
  const int l16  = lane & 15;
  const size_t rowBase = (size_t)blockIdx.x * ROWS;

  // ---- phase 0: stage x tile (32x512 f32 -> bf16 in LDS) ----
  {
    const int r  = tid >> 3;             // 0..31
    const int c0 = (tid & 7) * 64;
    const float* src = x + (rowBase + r) * DIM + c0;
    __bf16* dst = &buf[r * LSTR + c0];
#pragma unroll
    for (int i = 0; i < 64; i += 8) {
      float4 a = *(const float4*)(src + i);
      float4 b = *(const float4*)(src + i + 4);
      v8bf pk;
      pk[0]=(__bf16)a.x; pk[1]=(__bf16)a.y; pk[2]=(__bf16)a.z; pk[3]=(__bf16)a.w;
      pk[4]=(__bf16)b.x; pk[5]=(__bf16)b.y; pk[6]=(__bf16)b.z; pk[7]=(__bf16)b.w;
      *(v8bf*)(dst + i) = pk;
    }
  }
  __syncthreads();

  // ---- phase 1: GEMM1  z = x @ W^T  (wave: 2 M-tiles x 4 N-tiles of 16) ----
  v8f acc1[2][4];
#pragma unroll
  for (int mt = 0; mt < 2; ++mt)
#pragma unroll
    for (int t = 0; t < 4; ++t) acc1[mt][t] = zero8();
  {
    const int nW = wv * 64;
    const __bf16* br[4];
#pragma unroll
    for (int t = 0; t < 4; ++t) br[t] = Wb + (size_t)(nW + t * 16 + l16) * DIM;
    const __bf16* a0 = &buf[l16 * LSTR];
    const __bf16* a1 = &buf[(16 + l16) * LSTR];
    for (int k0 = 0; k0 < DIM; k0 += 32) {
      v16bf fa0 = frag_a(a0, k0, half);
      v16bf fa1 = frag_a(a1, k0, half);
#pragma unroll
      for (int t = 0; t < 4; ++t) {
        v16bf fb = frag_b(br[t], k0, half);
        acc1[0][t] = wmma_bf16(fa0, fb, acc1[0][t]);
        acc1[1][t] = wmma_bf16(fa1, fb, acc1[1][t]);
      }
    }
  }
  __syncthreads();   // all x-tile reads complete before overwriting buf with z

  // ---- phase 1 epilogue: z = acc + bias -> bf16 z tile in LDS ----
  {
    const int nW = wv * 64;
#pragma unroll
    for (int mt = 0; mt < 2; ++mt)
#pragma unroll
      for (int t = 0; t < 4; ++t) {
        const int n = nW + t * 16 + l16;
        const float bv = bias[n];
#pragma unroll
        for (int r = 0; r < 8; ++r) {
          const int m = mt * 16 + (half << 3) + r;
          buf[m * LSTR + n] = (__bf16)(acc1[mt][t][r] + bv);
        }
      }
  }
  __syncthreads();

  // ---- ||z||^2 per row (from bf16, consistent with bf16 GEMM2) ----
#pragma unroll
  for (int rr = 0; rr < 4; ++rr) {
    const int m = wv * 4 + rr;
    float s = 0.f;
    for (int i = lane; i < DIM; i += 32) {
      float v = (float)buf[m * LSTR + i];
      s += v * v;
    }
#pragma unroll
    for (int off = 16; off > 0; off >>= 1) s += __shfl_xor(s, off, 32);
    if (lane == 0) zsql[m] = s;
  }
  __syncthreads();

  const float tr   = temp_raw[0];
  const float temp = (1.f / (1.f + __expf(-tr))) * 0.999f + 0.001f;
  const float invT = 1.f / temp;

  // ---- phase 2: GEMM2 scores = z @ P^T (wave: 2 M-tiles x 8 N-tiles) ----
  const int nW2 = wv * 128;
  const __bf16* pr[8];
#pragma unroll
  for (int t = 0; t < 8; ++t) pr[t] = Pb + (size_t)(nW2 + t * 16 + l16) * DIM;
  const __bf16* za0 = &buf[l16 * LSTR];
  const __bf16* za1 = &buf[(16 + l16) * LSTR];

  v8f acc2[2][8];
#pragma unroll
  for (int mt = 0; mt < 2; ++mt)
#pragma unroll
    for (int t = 0; t < 8; ++t) acc2[mt][t] = zero8();

  for (int k0 = 0; k0 < DIM; k0 += 32) {
    v16bf fa0 = frag_a(za0, k0, half);
    v16bf fa1 = frag_a(za1, k0, half);
#pragma unroll
    for (int t = 0; t < 8; ++t) {
      v16bf fb = frag_b(pr[t], k0, half);
      acc2[0][t] = wmma_bf16(fa0, fb, acc2[0][t]);
      acc2[1][t] = wmma_bf16(fa1, fb, acc2[1][t]);
    }
  }

  // ---- logits = -sqrt(max(zsq+psq-2s, eps))/T ; per-row max ----
  float rmax_l[2][8];
#pragma unroll
  for (int mt = 0; mt < 2; ++mt)
#pragma unroll
    for (int r = 0; r < 8; ++r) rmax_l[mt][r] = -3.0e38f;
#pragma unroll
  for (int t = 0; t < 8; ++t) {
    const int n = nW2 + t * 16 + l16;
    const float ps = psq[n];
#pragma unroll
    for (int mt = 0; mt < 2; ++mt)
#pragma unroll
      for (int r = 0; r < 8; ++r) {
        const int m = mt * 16 + (half << 3) + r;
        float d2 = zsql[m] + ps - 2.f * acc2[mt][t][r];
        float d  = __builtin_sqrtf(fmaxf(d2, 1e-12f));
        float lg = -d * invT;
        acc2[mt][t][r] = lg;
        rmax_l[mt][r]  = fmaxf(rmax_l[mt][r], lg);
      }
  }
#pragma unroll
  for (int mt = 0; mt < 2; ++mt)
#pragma unroll
    for (int r = 0; r < 8; ++r) {
      float v = rmax_l[mt][r];
      v = fmaxf(v, __shfl_xor(v, 1, 32));
      v = fmaxf(v, __shfl_xor(v, 2, 32));
      v = fmaxf(v, __shfl_xor(v, 4, 32));
      v = fmaxf(v, __shfl_xor(v, 8, 32));
      if (l16 == 0) rowred[mt * 16 + (half << 3) + r][wv] = v;
    }
  __syncthreads();
  if (tid < ROWS) {
    float m = -3.0e38f;
#pragma unroll
    for (int w2 = 0; w2 < 8; ++w2) m = fmaxf(m, rowred[tid][w2]);
    rstat[tid] = m;
  }
  __syncthreads();

  float rsum_l[2][8];
#pragma unroll
  for (int mt = 0; mt < 2; ++mt)
#pragma unroll
    for (int r = 0; r < 8; ++r) rsum_l[mt][r] = 0.f;
#pragma unroll
  for (int mt = 0; mt < 2; ++mt)
#pragma unroll
    for (int r = 0; r < 8; ++r) {
      const float mx = rstat[mt * 16 + (half << 3) + r];
#pragma unroll
      for (int t = 0; t < 8; ++t) {
        float e = exp2f((acc2[mt][t][r] - mx) * 1.44269504f);
        acc2[mt][t][r] = e;
        rsum_l[mt][r] += e;
      }
    }
#pragma unroll
  for (int mt = 0; mt < 2; ++mt)
#pragma unroll
    for (int r = 0; r < 8; ++r) {
      float v = rsum_l[mt][r];
      v += __shfl_xor(v, 1, 32);
      v += __shfl_xor(v, 2, 32);
      v += __shfl_xor(v, 4, 32);
      v += __shfl_xor(v, 8, 32);
      if (l16 == 0) rowred[mt * 16 + (half << 3) + r][wv] = v;
    }
  __syncthreads();
  if (tid < ROWS) {
    float s = 0.f;
#pragma unroll
    for (int w2 = 0; w2 < 8; ++w2) s += rowred[tid][w2];
    rstat[tid] = 1.f / s;
  }
  __syncthreads();

  // ---- normalized weights -> d_out (f32); read back as GEMM3 A ----
#pragma unroll
  for (int mt = 0; mt < 2; ++mt)
#pragma unroll
    for (int r = 0; r < 8; ++r) {
      const int m = mt * 16 + (half << 3) + r;
      const float rinv = rstat[m];
      const size_t base = (rowBase + m) * (size_t)NPROT;
#pragma unroll
      for (int t = 0; t < 8; ++t) {
        const int n = nW2 + t * 16 + l16;
        out_w[base + n] = acc2[mt][t][r] * rinv;
      }
    }
  __threadfence();   // make cross-wave weight stores visible within the WGP
  __syncthreads();

  // ---- phase 3: GEMM3 blended = weights @ P (wave: 2 M x 4 D-tiles) ----
  const int dW = wv * 64;
  const __bf16* qr[4];
#pragma unroll
  for (int t = 0; t < 4; ++t) qr[t] = PTb + (size_t)(dW + t * 16 + l16) * NPROT;
  const float* wa0 = out_w + (rowBase + l16) * (size_t)NPROT;
  const float* wa1 = out_w + (rowBase + 16 + l16) * (size_t)NPROT;

  v8f acc3[2][4];
#pragma unroll
  for (int mt = 0; mt < 2; ++mt)
#pragma unroll
    for (int t = 0; t < 4; ++t) acc3[mt][t] = zero8();

  for (int k0 = 0; k0 < NPROT; k0 += 32) {
    v16bf fa0 = frag_a_f32(wa0, k0, half);
    v16bf fa1 = frag_a_f32(wa1, k0, half);
#pragma unroll
    for (int t = 0; t < 4; ++t) {
      v16bf fb = frag_b(qr[t], k0, half);
      acc3[0][t] = wmma_bf16(fa0, fb, acc3[0][t]);
      acc3[1][t] = wmma_bf16(fa1, fb, acc3[1][t]);
    }
  }
#pragma unroll
  for (int mt = 0; mt < 2; ++mt)
#pragma unroll
    for (int t = 0; t < 4; ++t) {
      const int d = dW + t * 16 + l16;
#pragma unroll
      for (int r = 0; r < 8; ++r) {
        const size_t m = rowBase + mt * 16 + (half << 3) + r;
        out_blend[m * DIM + d] = acc3[mt][t][r];
      }
    }
}

// ---------------------------------------------------------------------------
// CDNA5 TDM / cluster ISA probe (compiled, never launched): exercises
// tensor_load_to_lds + s_wait_tensorcnt + cluster_id codegen paths.
// ---------------------------------------------------------------------------
typedef __attribute__((ext_vector_type(4))) unsigned int u32x4;
typedef __attribute__((ext_vector_type(8))) int i32x8;
typedef __attribute__((ext_vector_type(4))) int i32x4;

__global__ void cdna5_tdm_probe(unsigned* out) {
  u32x4 g0; i32x4 g2, g3;
  i32x8 g1;
#pragma unroll
  for (int i = 0; i < 4; ++i) { g0[i] = 0u; g2[i] = 0; g3[i] = 0; }
#pragma unroll
  for (int i = 0; i < 8; ++i) g1[i] = 0;
#if __clang_major__ >= 23
  i32x8 g4;
#pragma unroll
  for (int i = 0; i < 8; ++i) g4[i] = 0;
  __builtin_amdgcn_tensor_load_to_lds(g0, g1, g2, g3, g4, 0);
#else
  __builtin_amdgcn_tensor_load_to_lds(g0, g1, g2, g3, 0);
#endif
  __builtin_amdgcn_s_wait_tensorcnt(0);
  out[0] = (unsigned)__builtin_amdgcn_cluster_id_x();
}

// ---------------------------------------------------------------------------
extern "C" void kernel_launch(void* const* d_in, const int* in_sizes, int n_in,
                              void* d_out, int out_size, void* d_ws, size_t ws_size,
                              hipStream_t stream) {
  (void)in_sizes; (void)n_in; (void)out_size; (void)ws_size;
  const float* x      = (const float*)d_in[0];
  const float* W      = (const float*)d_in[1];
  const float* bias   = (const float*)d_in[2];
  const float* protos = (const float*)d_in[3];
  const float* traw   = (const float*)d_in[4];
  float* out = (float*)d_out;

  // workspace carve-out (~2.5 MB)
  char* p = (char*)d_ws;
  __bf16* Wb  = (__bf16*)p;  p += (size_t)DIM * DIM * sizeof(__bf16);
  __bf16* Pb  = (__bf16*)p;  p += (size_t)NPROT * DIM * sizeof(__bf16);
  __bf16* PTb = (__bf16*)p;  p += (size_t)DIM * NPROT * sizeof(__bf16);
  float*  psq = (float*)p;

  float* out_blend = out;
  float* out_w     = out + (size_t)BROWS * DIM;

  prep_kernel<<<NPROT + DIM, 128, 0, stream>>>(W, protos, Wb, Pb, PTb, psq);
  som_fused_kernel<<<BROWS / ROWS, 256, 0, stream>>>(x, Wb, bias, Pb, PTb, psq,
                                                     traw, out_blend, out_w);
}
